// SpikeDrivenTransformerV2_79216376807554
// MI455X (gfx1250) — matrix operand order repaired and checked by reading, not compiled, not run on hardware
//
#include <hip/hip_runtime.h>

// ---- problem dims ----
#define T_     4
#define B_     16
#define CIN_   2
#define V_     128
#define E_     512
#define H_     8
#define DEPTH_ 6
#define NC_    11
#define HID_   (4 * E_)
#define EPS_   1e-5f

typedef __attribute__((ext_vector_type(16))) _Float16 v16h;
typedef __attribute__((ext_vector_type(8)))  _Float16 v8h;
typedef __attribute__((ext_vector_type(8)))  float    v8f;

// ---------------------------------------------------------------------------
// WMMA GEMM, activations channel-fastest:
//   Y[tb, v, o] = scale[o] * (sum_c S[tb, v, c] * W[o, c]) + bias[o] (+ resid)
// A-operand = spike tile (M = v, K = c)   -> two b128 loads/lane, contiguous
// B-operand = W^T tile   (K = c, N = o)   -> two b128 loads/lane, contiguous
// No LDS, no barriers: pure global_load + v_wmma_f32_16x16x32_f16, K unroll x2.
// grid = (O/64, V_/16, TB), block = 128 (4 independent wave32 waves).
// ---------------------------------------------------------------------------
__global__ __launch_bounds__(128)
void gemm_bn_wmma(const _Float16* __restrict__ W,   // [O, C] row-major
                  const _Float16* __restrict__ S,   // [TB, V_, C] f16 spikes
                  const float* __restrict__ scale,  // [O]
                  const float* __restrict__ bias,   // [O]
                  const float* __restrict__ resid,  // [TB, V_, O] or null
                  float* __restrict__ Y,            // [TB, V_, O]
                  int O, int C)
{
    const int lane = threadIdx.x & 31;     // wave32 (gfx1250)
    const int wave = threadIdx.x >> 5;
    const int tb   = blockIdx.z;
    const int v0   = blockIdx.y * 16;
    const int o0   = blockIdx.x * 64 + wave * 16;

    // A: ISA 16-bit A 16x32 layout: lane l -> M = l&15,
    //    K in {base..base+7} U {base+16..base+23}, base = 8*(l>>4)
    const _Float16* Arow = S + ((size_t)tb * V_ + v0 + (lane & 15)) * C;
    const int abase = (lane >> 4) * 8;
    // B: lane l -> N = l&15, K in {base..base+7} U {base+16..base+23}, base = 8*(l>>4)
    const _Float16* Brow = W + (size_t)(o0 + (lane & 15)) * C + (lane >> 4) * 8;

    v8f acc = {};
    for (int k0 = 0; k0 < C; k0 += 64) {
        // speculative prefetch of the next A K-slab (dropped if past end)
        __builtin_prefetch(Arow + k0 + 64, 0, 1);
        #pragma unroll
        for (int u = 0; u < 2; ++u) {
            const int k = k0 + u * 32;
            v8h alo = *(const v8h*)(Arow + k + abase);
            v8h ahi = *(const v8h*)(Arow + k + abase + 16);
            v16h a;
            #pragma unroll
            for (int i = 0; i < 8; ++i) { a[i] = alo[i]; a[i + 8] = ahi[i]; }
            v8h blo = *(const v8h*)(Brow + k);
            v8h bhi = *(const v8h*)(Brow + k + 16);
            v16h b;
            #pragma unroll
            for (int i = 0; i < 8; ++i) { b[i] = blo[i]; b[i + 8] = bhi[i]; }
            acc = __builtin_amdgcn_wmma_f32_16x16x32_f16(
                      false, a, false, b, (short)0, acc, false, false);
        }
    }

    // ---- epilogue: fused BN scale/bias (+ residual) ----
    // D layout: lane l, VGPR r -> M(v) = r + 8*(l>>4), N(o) = l&15
    const int o  = o0 + (lane & 15);
    const float sc = scale[o], bi = bias[o];
    #pragma unroll
    for (int r = 0; r < 8; ++r) {
        const int v = v0 + (lane >> 4) * 8 + r;
        float y = acc[r] * sc + bi;
        const size_t idx = ((size_t)tb * V_ + v) * O + o;
        if (resid) y += resid[idx];
        Y[idx] = y;
    }
}

// ---------------------------------------------------------------------------
// LIF scan over T (decay_input, tau=2, hard reset); emits f16 spikes {0,1}.
// Layout-agnostic elementwise scan: x f32 [T, M], s f16 [T, M].
// ---------------------------------------------------------------------------
__global__ void lif_f16_kernel(const float* __restrict__ x,
                               _Float16* __restrict__ s, float vth, int M)
{
    int i = blockIdx.x * blockDim.x + threadIdx.x;
    if (i >= M) return;
    float v = 0.f;
    #pragma unroll
    for (int t = 0; t < T_; ++t) {
        float xt = x[(size_t)t * M + i];
        v = 0.5f * (v + xt);                     // v += (x - v)/tau, tau = 2
        float sp = (v - vth >= 0.f) ? 1.f : 0.f;
        s[(size_t)t * M + i] = (_Float16)sp;
        v *= (1.f - sp);                         // hard reset
    }
}

__global__ void lif_f32_kernel(const float* __restrict__ x,
                               float* __restrict__ s, float vth, int M)
{
    int i = blockIdx.x * blockDim.x + threadIdx.x;
    if (i >= M) return;
    float v = 0.f;
    #pragma unroll
    for (int t = 0; t < T_; ++t) {
        float xt = x[(size_t)t * M + i];
        v = 0.5f * (v + xt);
        float sp = (v - vth >= 0.f) ? 1.f : 0.f;
        s[(size_t)t * M + i] = sp;
        v *= (1.f - sp);
    }
}

// ---------------------------------------------------------------------------
// Patch embed: h[tb, v, o] = bn(conv1x1(x, pe_w), pe_bn)
// x: [T,B,CIN_,V_] (given layout), h: [TB, V_, E] (channel-fastest)
// ---------------------------------------------------------------------------
__global__ void patch_embed_kernel(const float* __restrict__ x,
                                   const float* __restrict__ w,
                                   const float* __restrict__ bnp,
                                   float* __restrict__ h, int total)
{
    int i = blockIdx.x * blockDim.x + threadIdx.x;
    if (i >= total) return;
    int o  = i % E_;
    int v  = (i / E_) % V_;
    int tb = i / (E_ * V_);
    const float* xb = x + (size_t)tb * CIN_ * V_;
    float acc = w[o * CIN_ + 0] * xb[v] + w[o * CIN_ + 1] * xb[V_ + v];
    float g = bnp[o], bb = bnp[E_ + o], m = bnp[2 * E_ + o], var = bnp[3 * E_ + o];
    float sc = g * rsqrtf(var + EPS_);
    h[i] = (acc - m) * sc + bb;
}

// bn2(bn1(z)) -> z*scale + bias ;  bnp: [2][4][C] (g,b,m,v)
__global__ void fuse_bn2_kernel(const float* __restrict__ bnp,
                                float* __restrict__ scale,
                                float* __restrict__ bias, int C)
{
    int c = blockIdx.x * blockDim.x + threadIdx.x;
    if (c >= C) return;
    float g1 = bnp[c], b1 = bnp[C + c], m1 = bnp[2 * C + c], v1 = bnp[3 * C + c];
    const float* q = bnp + 4 * C;
    float g2 = q[c], b2 = q[C + c], m2 = q[2 * C + c], v2 = q[3 * C + c];
    float s1 = g1 * rsqrtf(v1 + EPS_), t1 = b1 - m1 * s1;
    float s2 = g2 * rsqrtf(v2 + EPS_), t2 = b2 - m2 * s2;
    scale[c] = s1 * s2;
    bias[c]  = t1 * s2 + t2;
}

// bn(Wx + fb) -> (Wx)*scale + bias ;  bnp: [4][C]
__global__ void fuse_bn_bias_kernel(const float* __restrict__ bnp,
                                    const float* __restrict__ fb,
                                    float* __restrict__ scale,
                                    float* __restrict__ bias, int C)
{
    int c = blockIdx.x * blockDim.x + threadIdx.x;
    if (c >= C) return;
    float g = bnp[c], b = bnp[C + c], m = bnp[2 * C + c], v = bnp[3 * C + c];
    float s = g * rsqrtf(v + EPS_);
    scale[c] = s;
    bias[c]  = (fb[c] - m) * s + b;
}

__global__ void cvt_f16_kernel(const float* __restrict__ src,
                               _Float16* __restrict__ dst, int n)
{
    int i = blockIdx.x * blockDim.x + threadIdx.x;
    if (i < n) dst[i] = (_Float16)src[i];
}

// kv[tb,h,c,d] = sum_n k[tb, n, h*64+c] * v[tb, n, h*64+d]   (spikes [TB,V_,E])
__global__ void kv_kernel(const _Float16* __restrict__ ks,
                          const _Float16* __restrict__ vs,
                          float* __restrict__ kv, int total)
{
    int i = blockIdx.x * blockDim.x + threadIdx.x;
    if (i >= total) return;
    int d  = i & 63;
    int c  = (i >> 6) & 63;
    int hh = (i >> 12) & 7;
    int tb = i >> 15;
    const _Float16* kb = ks + (size_t)tb * V_ * E_ + hh * 64 + c;
    const _Float16* vb = vs + (size_t)tb * V_ * E_ + hh * 64 + d;
    float acc = 0.f;
    #pragma unroll 8
    for (int n = 0; n < V_; ++n)
        acc += (float)kb[(size_t)n * E_] * (float)vb[(size_t)n * E_];
    kv[i] = acc;
}

// o[tb, n, h*64+d] = 0.125 * sum_c q[tb, n, h*64+c] * kv[tb,h,c,d]
__global__ void attn_o_kernel(const _Float16* __restrict__ qs,
                              const float* __restrict__ kv,
                              float* __restrict__ o, int total)
{
    int i = blockIdx.x * blockDim.x + threadIdx.x;
    if (i >= total) return;
    int ch = i & (E_ - 1);
    int n  = (i >> 9) & (V_ - 1);
    int tb = i >> 16;
    int hh = ch >> 6, d = ch & 63;
    const _Float16* qb = qs + ((size_t)tb * V_ + n) * E_ + hh * 64;   // contiguous over c
    const float* kvb   = kv + (((size_t)tb * H_ + hh) * 64) * 64 + d; // stride 64 over c
    float acc = 0.f;
    #pragma unroll 8
    for (int c = 0; c < 64; ++c) acc += (float)qb[c] * kvb[(size_t)c * 64];
    o[i] = acc * 0.125f;
}

// m[tb, e] = mean_v h[tb, v, e]   (h channel-fastest -> coalesced across threads)
__global__ void mean_v_kernel(const float* __restrict__ h,
                              float* __restrict__ m, int total)
{
    int i = blockIdx.x * blockDim.x + threadIdx.x;
    if (i >= total) return;
    int e  = i % E_;
    int tb = i / E_;
    const float* p = h + (size_t)tb * V_ * E_ + e;
    float acc = 0.f;
    #pragma unroll 8
    for (int v = 0; v < V_; ++v) acc += p[(size_t)v * E_];
    m[i] = acc * (1.f / V_);
}

// out[b,nc] = mean_t( s[t,b,:] . head_w[nc,:] ) + head_b[nc]
__global__ void head_kernel(const float* __restrict__ s,
                            const float* __restrict__ w,
                            const float* __restrict__ hb,
                            float* __restrict__ out, int total)
{
    int i = blockIdx.x * blockDim.x + threadIdx.x;
    if (i >= total) return;
    int nc = i % NC_, b = i / NC_;
    const float* wp = w + (size_t)nc * E_;
    float acc = 0.f;
    for (int t = 0; t < T_; ++t) {
        const float* sp = s + ((size_t)t * B_ + b) * E_;
        for (int e = 0; e < E_; ++e) acc += sp[e] * wp[e];
    }
    out[(size_t)b * NC_ + nc] = acc * (1.f / T_) + hb[nc];
}

// ---------------------------------------------------------------------------
extern "C" void kernel_launch(void* const* d_in, const int* in_sizes, int n_in,
                              void* d_out, int out_size, void* d_ws, size_t ws_size,
                              hipStream_t stream)
{
    (void)in_sizes; (void)n_in; (void)out_size; (void)ws_size;
    const float* x       = (const float*)d_in[0];
    const float* pe_w    = (const float*)d_in[1];
    const float* pe_bn   = (const float*)d_in[2];
    const float* qkvp_w  = (const float*)d_in[3];
    const float* qkvp_bn = (const float*)d_in[4];
    const float* fc1_w   = (const float*)d_in[5];
    const float* fc1_b   = (const float*)d_in[6];
    const float* fc1_bn  = (const float*)d_in[7];
    const float* fc2_w   = (const float*)d_in[8];
    const float* fc2_b   = (const float*)d_in[9];
    const float* fc2_bn  = (const float*)d_in[10];
    const float* head_w  = (const float*)d_in[11];
    const float* head_b  = (const float*)d_in[12];
    float* out = (float*)d_out;

    const int TB = T_ * B_;
    char* p = (char*)d_ws;
    auto alloc = [&](size_t bytes) -> char* {
        char* r = p; p += (bytes + 255) & ~(size_t)255; return r;
    };
    // All activations live as [TB, V_, Ch] (channel-fastest).
    float*    h    = (float*)   alloc((size_t)TB * V_ * E_ * 4);    // residual stream
    float*    z    = (float*)   alloc((size_t)TB * V_ * E_ * 4);    // branch/attn-o temp
    _Float16* s0   = (_Float16*)alloc((size_t)TB * V_ * E_ * 2);    // spike scratch
    _Float16* qs   = (_Float16*)alloc((size_t)TB * V_ * E_ * 2);
    _Float16* ks   = (_Float16*)alloc((size_t)TB * V_ * E_ * 2);
    _Float16* vs   = (_Float16*)alloc((size_t)TB * V_ * E_ * 2);
    float*    kvb  = (float*)   alloc((size_t)TB * H_ * 64 * 64 * 4);
    float*    z1   = (float*)   alloc((size_t)TB * V_ * HID_ * 4);
    _Float16* s1   = (_Float16*)alloc((size_t)TB * V_ * HID_ * 2);
    _Float16* wbuf = (_Float16*)alloc((size_t)HID_ * E_ * 2);       // per-GEMM f16 weights
    float*    scb  = (float*)   alloc((size_t)HID_ * 4);
    float*    bib  = (float*)   alloc((size_t)HID_ * 4);
    float*    mbuf = (float*)   alloc((size_t)T_ * B_ * E_ * 4);
    float*    shd  = (float*)   alloc((size_t)T_ * B_ * E_ * 4);

    const int BS = 256;
    auto g1 = [](int n) { return dim3((unsigned)((n + 255) / 256)); };

    const int totPE = TB * V_ * E_;       // 4 Mi elements
    const int ME    = B_ * V_ * E_;       // per-timestep element count (E path)
    const int MH    = B_ * V_ * HID_;     // per-timestep element count (HID path)

    patch_embed_kernel<<<g1(totPE), BS, 0, stream>>>(x, pe_w, pe_bn, h, totPE);

    for (int d = 0; d < DEPTH_; ++d) {
        // ---------- attention ----------
        lif_f16_kernel<<<g1(ME), BS, 0, stream>>>(h, s0, 1.0f, ME);
        _Float16* sp3[3] = {qs, ks, vs};
        for (int i = 0; i < 3; ++i) {
            cvt_f16_kernel<<<g1(E_ * E_), BS, 0, stream>>>(
                qkvp_w + (size_t)(d * 4 + i) * E_ * E_, wbuf, E_ * E_);
            fuse_bn2_kernel<<<g1(E_), BS, 0, stream>>>(
                qkvp_bn + (size_t)(d * 4 + i) * 8 * E_, scb, bib, E_);
            gemm_bn_wmma<<<dim3(E_ / 64, V_ / 16, TB), 128, 0, stream>>>(
                wbuf, s0, scb, bib, nullptr, z, E_, E_);
            lif_f16_kernel<<<g1(ME), BS, 0, stream>>>(z, sp3[i], 1.0f, ME);
        }
        const int totKV = TB * H_ * 64 * 64;
        kv_kernel<<<g1(totKV), BS, 0, stream>>>(ks, vs, kvb, totKV);
        attn_o_kernel<<<g1(totPE), BS, 0, stream>>>(qs, kvb, z, totPE);
        lif_f16_kernel<<<g1(ME), BS, 0, stream>>>(z, s0, 0.5f, ME);  // attn_lif (vth=0.5)
        cvt_f16_kernel<<<g1(E_ * E_), BS, 0, stream>>>(
            qkvp_w + (size_t)(d * 4 + 3) * E_ * E_, wbuf, E_ * E_);
        fuse_bn2_kernel<<<g1(E_), BS, 0, stream>>>(
            qkvp_bn + (size_t)(d * 4 + 3) * 8 * E_, scb, bib, E_);
        gemm_bn_wmma<<<dim3(E_ / 64, V_ / 16, TB), 128, 0, stream>>>(
            wbuf, s0, scb, bib, /*resid=*/h, /*Y=*/h, E_, E_);   // h += proj(...)

        // ---------- mlp ----------
        lif_f16_kernel<<<g1(ME), BS, 0, stream>>>(h, s0, 1.0f, ME);
        cvt_f16_kernel<<<g1(HID_ * E_), BS, 0, stream>>>(
            fc1_w + (size_t)d * HID_ * E_, wbuf, HID_ * E_);
        fuse_bn_bias_kernel<<<g1(HID_), BS, 0, stream>>>(
            fc1_bn + (size_t)d * 4 * HID_, fc1_b + (size_t)d * HID_, scb, bib, HID_);
        gemm_bn_wmma<<<dim3(HID_ / 64, V_ / 16, TB), 128, 0, stream>>>(
            wbuf, s0, scb, bib, nullptr, z1, HID_, E_);
        lif_f16_kernel<<<g1(MH), BS, 0, stream>>>(z1, s1, 1.0f, MH);
        cvt_f16_kernel<<<g1(E_ * HID_), BS, 0, stream>>>(
            fc2_w + (size_t)d * E_ * HID_, wbuf, E_ * HID_);
        fuse_bn_bias_kernel<<<g1(E_), BS, 0, stream>>>(
            fc2_bn + (size_t)d * 4 * E_, fc2_b + (size_t)d * E_, scb, bib, E_);
        gemm_bn_wmma<<<dim3(E_ / 64, V_ / 16, TB), 128, 0, stream>>>(
            wbuf, s1, scb, bib, /*resid=*/h, /*Y=*/h, E_, HID_);  // h += mlp(...)
    }

    const int totM = TB * E_;
    mean_v_kernel<<<g1(totM), BS, 0, stream>>>(h, mbuf, totM);
    lif_f32_kernel<<<g1(B_ * E_), BS, 0, stream>>>(mbuf, shd, 1.0f, B_ * E_);
    head_kernel<<<g1(B_ * NC_), BS, 0, stream>>>(shd, head_w, head_b, out, B_ * NC_);
}
